// SimpleDynamicBrainNet_71159018160619
// MI455X (gfx1250) — matrix operand order repaired and checked by reading, not compiled, not run on hardware
//
#include <hip/hip_runtime.h>
#include <hip/hip_bf16.h>

typedef __attribute__((ext_vector_type(16))) _Float16 v16h;
typedef __attribute__((ext_vector_type(8)))  float    v8f;

#define NODE_DIM 16
#define HID      64
#define NCLS     2
#define BB       16
#define TT       64
#define NN       400

// ---------------------------------------------------------------------------
// Kernel 1: per (b,t):
//   s[m]   = sum_n adj[b,t,n,m]                      (streams 640KB/block)
//   y[d]   = sum_m s[m] * x[b,t,m,d],  ssum = sum_m s[m]
//   feats[t,b,h] = (sum_d y[d]*Wp[d,h] + ssum*bp[h]) / N
// ---------------------------------------------------------------------------
__global__ __launch_bounds__(512)
void brainnet_feats_kernel(const float* __restrict__ x,
                           const float* __restrict__ adj,
                           const float* __restrict__ Wp,
                           const float* __restrict__ bp,
                           float* __restrict__ feats)
{
    __shared__ float s_lds[NN];
    __shared__ float part[4 * 16];
    __shared__ float spart[4];
    __shared__ float y_lds[16];
    __shared__ float ssum_lds;

    const int bt  = blockIdx.x;      // b*T + t  (matches adj/x layout)
    const int b   = bt / TT;
    const int t   = bt % TT;
    const int tid = threadIdx.x;

    const float* adj_bt = adj + (size_t)bt * NN * NN;
    const float* x_bt   = x   + (size_t)bt * NN * NODE_DIM;

    // Phase A: column sums (fully coalesced: lane = column, loop over rows)
    if (tid < NN) {
        float acc = 0.f;
        #pragma unroll 4
        for (int n = 0; n < NN; ++n)
            acc += adj_bt[(size_t)n * NN + tid];
        s_lds[tid] = acc;
    }
    __syncthreads();

    // Phase B: y[d] (4-way split reduction over m) and ssum
    if (tid < 64) {
        const int d = tid & 15, g = tid >> 4;
        float acc = 0.f;
        for (int m = g; m < NN; m += 4)
            acc += s_lds[m] * x_bt[m * NODE_DIM + d];
        part[g * 16 + d] = acc;
    } else if (tid < 68) {
        const int g = tid - 64;
        float acc = 0.f;
        for (int m = g; m < NN; m += 4) acc += s_lds[m];
        spart[g] = acc;
    }
    __syncthreads();
    if (tid < 16) y_lds[tid] = part[tid] + part[16 + tid] + part[32 + tid] + part[48 + tid];
    if (tid == 16) ssum_lds = spart[0] + spart[1] + spart[2] + spart[3];
    __syncthreads();

    // Phase C: project to HID, store feats in [T][B][HID] for GRU step access
    if (tid < HID) {
        float acc = ssum_lds * bp[tid];
        #pragma unroll
        for (int d = 0; d < NODE_DIM; ++d)
            acc += y_lds[d] * Wp[d * HID + tid];
        feats[(size_t)t * BB * HID + b * HID + tid] = acc * (1.0f / NN);
    }
}

// ---------------------------------------------------------------------------
// Kernel 2: GRU (T=64 sequential steps, batch 16, hidden 64) + classifier.
// Single workgroup, 4 waves. Per step, two matmuls (16x64)@(64x192) done
// with v_wmma_f32_16x16x32_f16: 12 N-tiles, K=64 as 2 chunks of 32.
// Wave w owns tiles {w, w+4, w+8} == r/z/n gates for hidden cols [16w,16w+16),
// so gate math and the h update are lane-local in the C/D layout.
// ---------------------------------------------------------------------------
__global__ __launch_bounds__(128)
void brainnet_gru_kernel(const float* __restrict__ feats,
                         const float* __restrict__ W_ih,
                         const float* __restrict__ W_hh,
                         const float* __restrict__ b_ih,
                         const float* __restrict__ b_hh,
                         const float* __restrict__ Wc,
                         const float* __restrict__ bc,
                         float* __restrict__ out)
{
    __shared__ __align__(32) _Float16 Wih_lds[3 * HID * HID];   // [192][64] row-major
    __shared__ __align__(32) _Float16 Whh_lds[3 * HID * HID];
    __shared__ __align__(32) _Float16 h_lds[2][BB * HID];       // double-buffered f16 h
    __shared__ float h32_lds[BB * HID];
    __shared__ float bih_lds[3 * HID];
    __shared__ float bhh_lds[3 * HID];

    const int tid   = threadIdx.x;
    const int wave  = tid >> 5;
    const int lane  = tid & 31;
    const int nlo   = lane & 15;
    const bool hi   = lane >= 16;
    const int mbase = hi ? 8 : 0;
    const int col   = wave * 16 + nlo;      // hidden column owned by this lane

    // Preload weights (f32 -> f16) and biases into LDS; zero h[buf 0]
    for (int i = tid; i < 3 * HID * HID; i += 128) {
        Wih_lds[i] = (_Float16)W_ih[i];
        Whh_lds[i] = (_Float16)W_hh[i];
    }
    for (int i = tid; i < 3 * HID; i += 128) {
        bih_lds[i] = b_ih[i];
        bhh_lds[i] = b_hh[i];
    }
    for (int i = tid; i < BB * HID; i += 128) h_lds[0][i] = (_Float16)0.f;
    __syncthreads();

    const float bir = bih_lds[0 * HID + col], bhr = bhh_lds[0 * HID + col];
    const float biz = bih_lds[1 * HID + col], bhz = bhh_lds[1 * HID + col];
    const float bin = bih_lds[2 * HID + col], bhn = bhh_lds[2 * HID + col];

    float hreg[8];
    #pragma unroll
    for (int i = 0; i < 8; ++i) hreg[i] = 0.f;

    for (int t = 0; t < TT; ++t) {
        // ---- A fragments (16x32 f16, contiguous k = 16*half + s) for feats_t (global) and h (LDS)
        union { v16h v; _Float16 e[16]; } af0, af1, ah0, ah1;
        {
            const int m  = nlo;                       // A row = batch index
            const int k0 = hi ? 16 : 0;
            const float*    fr = feats + (size_t)t * BB * HID + m * HID;
            const _Float16* hr = &h_lds[t & 1][m * HID];
            #pragma unroll
            for (int i = 0; i < 8; ++i) {
                af0.e[i]     = (_Float16)fr[k0 + i];
                af0.e[8 + i] = (_Float16)fr[8 + k0 + i];
                af1.e[i]     = (_Float16)fr[32 + k0 + i];
                af1.e[8 + i] = (_Float16)fr[40 + k0 + i];
                ah0.e[i]     = hr[k0 + i];
                ah0.e[8 + i] = hr[8 + k0 + i];
                ah1.e[i]     = hr[32 + k0 + i];
                ah1.e[8 + i] = hr[40 + k0 + i];
            }
        }

        // ---- 3 gate tiles per wave: gi and gh via chained WMMA over K chunks
        v8f acc_i[3], acc_h[3];
        #pragma unroll
        for (int g = 0; g < 3; ++g) {
            const int tile = wave + g * 4;            // g=0:r  g=1:z  g=2:n
            const int j    = tile * 16 + nlo;         // W row (output unit, 0..191)
            const int koff = hi ? 16 : 0;             // B layout: lane half -> K half
            v16h bi0 = *(const v16h*)&Wih_lds[j * HID + koff];
            v16h bi1 = *(const v16h*)&Wih_lds[j * HID + 32 + koff];
            v16h bh0 = *(const v16h*)&Whh_lds[j * HID + koff];
            v16h bh1 = *(const v16h*)&Whh_lds[j * HID + 32 + koff];
            v8f ci = {};
            ci = __builtin_amdgcn_wmma_f32_16x16x32_f16(false, af0.v, false, bi0,
                                                        (short)0, ci, false, false);
            ci = __builtin_amdgcn_wmma_f32_16x16x32_f16(false, af1.v, false, bi1,
                                                        (short)0, ci, false, false);
            v8f ch = {};
            ch = __builtin_amdgcn_wmma_f32_16x16x32_f16(false, ah0.v, false, bh0,
                                                        (short)0, ch, false, false);
            ch = __builtin_amdgcn_wmma_f32_16x16x32_f16(false, ah1.v, false, bh1,
                                                        (short)0, ch, false, false);
            acc_i[g] = ci;
            acc_h[g] = ch;
        }

        // ---- Gates + h update (lane-local: D layout lane=N col, vgpr idx=M row)
        union { v8f v; float f[8]; } gir, ghr, giz, ghz, gin, ghn;
        gir.v = acc_i[0]; ghr.v = acc_h[0];
        giz.v = acc_i[1]; ghz.v = acc_h[1];
        gin.v = acc_i[2]; ghn.v = acc_h[2];

        _Float16* hw = &h_lds[(t + 1) & 1][0];
        #pragma unroll
        for (int m = 0; m < 8; ++m) {
            const float r  = 1.f / (1.f + expf(-(gir.f[m] + bir + ghr.f[m] + bhr)));
            const float z  = 1.f / (1.f + expf(-(giz.f[m] + biz + ghz.f[m] + bhz)));
            const float n  = tanhf(gin.f[m] + bin + r * (ghn.f[m] + bhn));
            const float hv = (1.f - z) * n + z * hreg[m];
            hreg[m] = hv;
            hw[(m + mbase) * HID + col] = (_Float16)hv;
        }
        __syncthreads();   // publishes h for next step; guards double buffer
    }

    // ---- Classifier: logits = h_last @ Wc + bc  (16x64 @ 64x2)
    #pragma unroll
    for (int m = 0; m < 8; ++m) h32_lds[(m + mbase) * HID + col] = hreg[m];
    __syncthreads();
    if (tid < BB * NCLS) {
        const int b = tid >> 1, c = tid & 1;
        float acc = bc[c];
        for (int k = 0; k < HID; ++k)
            acc += h32_lds[b * HID + k] * Wc[k * NCLS + c];
        out[tid] = acc;
    }
}

// ---------------------------------------------------------------------------
extern "C" void kernel_launch(void* const* d_in, const int* in_sizes, int n_in,
                              void* d_out, int out_size, void* d_ws, size_t ws_size,
                              hipStream_t stream)
{
    const float* x    = (const float*)d_in[0];   // x_seq   (B,T,N,16)
    const float* adj  = (const float*)d_in[1];   // adj_seq (B,T,N,N)
    const float* Wp   = (const float*)d_in[2];   // (16,64)
    const float* bp   = (const float*)d_in[3];   // (64,)
    const float* W_ih = (const float*)d_in[4];   // (192,64)
    const float* W_hh = (const float*)d_in[5];   // (192,64)
    const float* b_ih = (const float*)d_in[6];   // (192,)
    const float* b_hh = (const float*)d_in[7];   // (192,)
    const float* Wc   = (const float*)d_in[8];   // (64,2)
    const float* bc   = (const float*)d_in[9];   // (2,)
    float* out   = (float*)d_out;                // (16,2)
    float* feats = (float*)d_ws;                 // [T][B][HID] f32 = 256 KB

    brainnet_feats_kernel<<<BB * TT, 512, 0, stream>>>(x, adj, Wp, bp, feats);
    brainnet_gru_kernel<<<1, 128, 0, stream>>>(feats, W_ih, W_hh, b_ih, b_hh,
                                               Wc, bc, out);
}
